// QLoRALinear_1726576853689
// MI455X (gfx1250) — matrix operand order repaired and checked by reading, not compiled, not run on hardware
//
#include <hip/hip_runtime.h>

typedef __attribute__((ext_vector_type(16))) _Float16 v16h;
typedef __attribute__((ext_vector_type(8)))  _Float16 v8h;
typedef __attribute__((ext_vector_type(8)))  float    v8f;
typedef __attribute__((ext_vector_type(4)))  float    v4f;
typedef __attribute__((ext_vector_type(4)))  int      v4i;

constexpr int kB   = 1024;   // batch
constexpr int kIn  = 4096;   // in features
constexpr int kOut = 4096;   // out features
constexpr int kR   = 16;     // lora rank
constexpr int kBM  = 128;    // block tile M
constexpr int kBN  = 128;    // block tile N
constexpr int kBK  = 32;     // K step (matches WMMA K=32)
constexpr int kLDA = 40;     // padded LDS row in halves (80B, 16B aligned)
constexpr int kLDB = 40;

__device__ __forceinline__ v16h cat8(v8h lo, v8h hi) {
  return __builtin_shufflevector(lo, hi, 0,1,2,3,4,5,6,7,8,9,10,11,12,13,14,15);
}

// ---------------------------------------------------------------------------
// t[b][r] = alpha * sum_i x[b,i] * lora_a[r,i]      (one wave per (b,r) pair)
// ---------------------------------------------------------------------------
__global__ void lora_t_kernel(const float* __restrict__ x,
                              const float* __restrict__ la,
                              const float* __restrict__ alpha,
                              float* __restrict__ t) {
  const int gwave = (blockIdx.x * blockDim.x + threadIdx.x) >> 5;
  const int lane  = threadIdx.x & 31;
  const int b = gwave >> 4;   // 0..1023
  const int r = gwave & 15;   // 0..15
  const float* xr = x  + (size_t)b * kIn;
  const float* ar = la + (size_t)r * kIn;
  float acc = 0.f;
  #pragma unroll 4
  for (int i = lane * 4; i < kIn; i += 32 * 4) {
    v4f xv = *(const v4f*)(xr + i);
    v4f av = *(const v4f*)(ar + i);
    acc += xv.x * av.x + xv.y * av.y + xv.z * av.z + xv.w * av.w;
  }
  #pragma unroll
  for (int off = 16; off > 0; off >>= 1) acc += __shfl_xor(acc, off, 32);
  if (lane == 0) t[b * kR + r] = alpha[0] * acc;
}

// ---------------------------------------------------------------------------
// Fused dequant(4-bit) + WMMA f16 GEMM + LoRA epilogue, double-buffered LDS.
// Block: 128x128 tile of C, 8 waves, each wave a 32x64 sub-tile (2x4 WMMA).
// ---------------------------------------------------------------------------
__global__ void __launch_bounds__(256, 1)
qlora_wmma_kernel(const float* __restrict__ x,
                  const int*   __restrict__ qw,
                  const float* __restrict__ scale,
                  const float* __restrict__ lb,
                  const float* __restrict__ t,
                  float* __restrict__ out) {
  __shared__ __align__(16) _Float16 As[2][kBM * kLDA];   // 2 x 10240 B
  __shared__ __align__(16) _Float16 Bs[2][kBN * kLDB];   // 2 x 10240 B

  const int tid    = threadIdx.x;
  const int lane   = tid & 31;
  const int wave   = tid >> 5;
  const int wm     = wave & 3;        // 4 waves along M (4*32 = 128)
  const int wn     = wave >> 2;       // 2 waves along N (2*64 = 128)
  const int half16 = lane >> 4;       // 0/1
  const int l16    = lane & 15;

  const int n0 = blockIdx.x * kBN;
  const int m0 = blockIdx.y * kBM;

  // staging roles
  const int xrow = tid >> 1;          // 0..127 : x tile row
  const int xh   = tid & 1;           // which 16-float half of the row
  const int oloc = tid >> 1;          // 0..127 : weight tile row (o)
  const int gk   = tid & 1;           // which 16-wide group along k

  const float* xsrc0 = x + (size_t)(m0 + xrow) * kIn + xh * 16;   // + k0
  const int    gbase = (n0 + oloc) * (kIn / 16) + gk;             // + (k0>>4)

  v8f acc[2][4];
  #pragma unroll
  for (int mi = 0; mi < 2; ++mi)
    #pragma unroll
    for (int ni = 0; ni < 4; ++ni) acc[mi][ni] = (v8f){};

  // ---- staging helper (stages tile at k0 into buffer `buf`) ----
  auto stage = [&](int buf, int k0) {
    // x tile: 128 x 32 f32 -> f16
    const float* src = xsrc0 + k0;
    v4f f0 = *(const v4f*)(src + 0);
    v4f f1 = *(const v4f*)(src + 4);
    v4f f2 = *(const v4f*)(src + 8);
    v4f f3 = *(const v4f*)(src + 12);
    v8h p0, p1;
    p0[0]=(_Float16)f0.x; p0[1]=(_Float16)f0.y; p0[2]=(_Float16)f0.z; p0[3]=(_Float16)f0.w;
    p0[4]=(_Float16)f1.x; p0[5]=(_Float16)f1.y; p0[6]=(_Float16)f1.z; p0[7]=(_Float16)f1.w;
    p1[0]=(_Float16)f2.x; p1[1]=(_Float16)f2.y; p1[2]=(_Float16)f2.z; p1[3]=(_Float16)f2.w;
    p1[4]=(_Float16)f3.x; p1[5]=(_Float16)f3.y; p1[6]=(_Float16)f3.z; p1[7]=(_Float16)f3.w;
    _Float16* xdst = &As[buf][xrow * kLDA + xh * 16];
    *(v8h*)(xdst + 0) = p0;
    *(v8h*)(xdst + 8) = p1;

    // weight tile: one full 16-value group per thread (dequant 4-bit -> f16)
    const int g = gbase + (k0 >> 4);
    v4i qa = *(const v4i*)(qw + (size_t)g * 8 + 0);
    v4i qb = *(const v4i*)(qw + (size_t)g * 8 + 4);
    float s  = scale[g];
    float ms = s * (2.0f / 15.0f);
    v8h h0, h1;
    h0[0]=(_Float16)((float)( qa.x        & 15) * ms - s);
    h0[1]=(_Float16)((float)((qa.x >>  4) & 15) * ms - s);
    h0[2]=(_Float16)((float)( qa.y        & 15) * ms - s);
    h0[3]=(_Float16)((float)((qa.y >>  4) & 15) * ms - s);
    h0[4]=(_Float16)((float)( qa.z        & 15) * ms - s);
    h0[5]=(_Float16)((float)((qa.z >>  4) & 15) * ms - s);
    h0[6]=(_Float16)((float)( qa.w        & 15) * ms - s);
    h0[7]=(_Float16)((float)((qa.w >>  4) & 15) * ms - s);
    h1[0]=(_Float16)((float)( qb.x        & 15) * ms - s);
    h1[1]=(_Float16)((float)((qb.x >>  4) & 15) * ms - s);
    h1[2]=(_Float16)((float)( qb.y        & 15) * ms - s);
    h1[3]=(_Float16)((float)((qb.y >>  4) & 15) * ms - s);
    h1[4]=(_Float16)((float)( qb.z        & 15) * ms - s);
    h1[5]=(_Float16)((float)((qb.z >>  4) & 15) * ms - s);
    h1[6]=(_Float16)((float)( qb.w        & 15) * ms - s);
    h1[7]=(_Float16)((float)((qb.w >>  4) & 15) * ms - s);
    _Float16* wdst = &Bs[buf][oloc * kLDB + gk * 16];
    *(v8h*)(wdst + 0) = h0;
    *(v8h*)(wdst + 8) = h1;
  };

  // prologue: stage first tile into buffer 0
  stage(0, 0);

  int buf = 0;
  for (int k0 = 0; k0 < kIn; k0 += kBK) {
    __syncthreads();                       // staging of `buf` complete; prev reads done
    if (k0 + kBK < kIn) {
      stage(buf ^ 1, k0 + kBK);            // overlap next-tile staging with WMMA
      __builtin_prefetch(xsrc0 + k0 + 2 * kBK, 0, 1);
      __builtin_prefetch(qw + (size_t)(gbase + ((k0 >> 4) + 4)) * 8, 0, 1);
    }

    // ---- fragments (documented CDNA5 VGPR layouts) ----
    // A: lane holds row M=l16, K = {8*half16..+7} and {16+8*half16..+7}
    const _Float16* Ab = As[buf];
    const int ar0 = (wm * 32 + l16) * kLDA + half16 * 8;
    v16h a0 = cat8(*(const v8h*)&Ab[ar0],             *(const v8h*)&Ab[ar0 + 16]);
    const int ar1 = ar0 + 16 * kLDA;
    v16h a1 = cat8(*(const v8h*)&Ab[ar1],             *(const v8h*)&Ab[ar1 + 16]);
    // B: lane holds column N=l16, K = {8*half16..+7} and {16+8*half16..+7}   (Bs is [n][k])
    const _Float16* Bb = Bs[buf];
    v16h bf[4];
    #pragma unroll
    for (int ni = 0; ni < 4; ++ni) {
      const int br = (wn * 64 + ni * 16 + l16) * kLDB + half16 * 8;
      bf[ni] = cat8(*(const v8h*)&Bb[br], *(const v8h*)&Bb[br + 16]);
    }

    #pragma unroll
    for (int ni = 0; ni < 4; ++ni) {
      acc[0][ni] = __builtin_amdgcn_wmma_f32_16x16x32_f16(false, a0, false, bf[ni],
                                                          (short)0, acc[0][ni], false, false);
      acc[1][ni] = __builtin_amdgcn_wmma_f32_16x16x32_f16(false, a1, false, bf[ni],
                                                          (short)0, acc[1][ni], false, false);
    }
    buf ^= 1;
  }

  // ---- epilogue: add LoRA (t[row,:] . lora_b[col,:]) and store f32 ----
  int   cols[4];
  v4f   lbv[4][4];
  #pragma unroll
  for (int ni = 0; ni < 4; ++ni) {
    cols[ni] = n0 + wn * 64 + ni * 16 + l16;
    #pragma unroll
    for (int j = 0; j < 4; ++j)
      lbv[ni][j] = *(const v4f*)(lb + (size_t)cols[ni] * kR + 4 * j);
  }
  const int rowbase = m0 + wm * 32 + half16 * 8;
  #pragma unroll
  for (int v = 0; v < 8; ++v) {
    const int row0 = rowbase + v;        // mi = 0
    const int row1 = row0 + 16;          // mi = 1
    v4f t0[4], t1[4];
    #pragma unroll
    for (int j = 0; j < 4; ++j) {
      t0[j] = *(const v4f*)(t + (size_t)row0 * kR + 4 * j);
      t1[j] = *(const v4f*)(t + (size_t)row1 * kR + 4 * j);
    }
    #pragma unroll
    for (int ni = 0; ni < 4; ++ni) {
      float d0 = 0.f, d1 = 0.f;
      #pragma unroll
      for (int j = 0; j < 4; ++j) {
        d0 += t0[j].x*lbv[ni][j].x + t0[j].y*lbv[ni][j].y +
              t0[j].z*lbv[ni][j].z + t0[j].w*lbv[ni][j].w;
        d1 += t1[j].x*lbv[ni][j].x + t1[j].y*lbv[ni][j].y +
              t1[j].z*lbv[ni][j].z + t1[j].w*lbv[ni][j].w;
      }
      out[(size_t)row0 * kOut + cols[ni]] = acc[0][ni][v] + d0;
      out[(size_t)row1 * kOut + cols[ni]] = acc[1][ni][v] + d1;
    }
  }
}

extern "C" void kernel_launch(void* const* d_in, const int* in_sizes, int n_in,
                              void* d_out, int out_size, void* d_ws, size_t ws_size,
                              hipStream_t stream) {
  (void)in_sizes; (void)n_in; (void)out_size; (void)ws_size;
  const float* x     = (const float*)d_in[0];
  const int*   qw    = (const int*)d_in[1];
  const float* scale = (const float*)d_in[2];
  const float* la    = (const float*)d_in[3];
  const float* lbp   = (const float*)d_in[4];
  const float* alpha = (const float*)d_in[5];
  float* out = (float*)d_out;
  float* t   = (float*)d_ws;   // 1024 x 16 f32 = 64 KB scratch

  // t = alpha * x @ lora_a^T   (16384 waves, 8 waves/block)
  lora_t_kernel<<<(kB * kR) / 8, 256, 0, stream>>>(x, la, alpha, t);

  dim3 grid(kOut / kBN, kB / kBM);   // (32, 8)
  qlora_wmma_kernel<<<grid, 256, 0, stream>>>(x, qw, scale, lbp, t, out);
}
